// STMetaLearner_update_54322746359871
// MI455X (gfx1250) — matrix operand order repaired and checked by reading, not compiled, not run on hardware
//
#include <hip/hip_runtime.h>
#include <math.h>

// ---------------- problem constants ----------------
constexpr int B_    = 32;
constexpr int N_    = 256;
constexpr int HIS   = 14;
constexpr int MSG   = 2;
constexpr int HID   = 64;
constexpr int META  = 32;
constexpr int HEADS = 3;
constexpr int E_    = 262144;
constexpr int T1    = HIS - 2;       // 12
constexpr int T2    = T1 - 2;        // 10
constexpr int BN    = B_ * N_;       // 8192
constexpr int EP    = E_ + BN;       // edges + self loops = 270336
constexpr int GM    = T1 * BN;       // 98304 GEMM rows
constexpr int GN    = HEADS * META;  // 96
constexpr float NEG_SLOPE = 0.2f;

// ---------------- WMMA fragment types ----------------
typedef __attribute__((ext_vector_type(16))) _Float16 v16h;
typedef __attribute__((ext_vector_type(8)))  float    v8f;

union HFrag { v16h v; _Float16 h[16]; };
union CFrag { v8f  v; float    f[8];  };

__device__ __forceinline__ float sigmoidf_(float x) { return 1.f / (1.f + expf(-x)); }

__device__ __forceinline__ void atomicMaxFloat(float* addr, float val) {
    // IEEE trick: int max for non-negative, uint min for negative; init = -inf
    if (val >= 0.f) atomicMax((int*)addr, __float_as_int(val));
    else            atomicMin((unsigned int*)addr, __float_as_uint(val));
}

// load 8 contiguous f32 and pack into 8 consecutive f16 fragment slots
__device__ __forceinline__ void load_run8(const float* __restrict__ p, _Float16* dst) {
    float4 q0 = *(const float4*)(p);
    float4 q1 = *(const float4*)(p + 4);
    dst[0] = (_Float16)q0.x; dst[1] = (_Float16)q0.y;
    dst[2] = (_Float16)q0.z; dst[3] = (_Float16)q0.w;
    dst[4] = (_Float16)q1.x; dst[5] = (_Float16)q1.y;
    dst[6] = (_Float16)q1.z; dst[7] = (_Float16)q1.w;
}

// ---------------- K1: TimeBlock1 (gated temporal conv) -> xs (t, node, c) ----------------
__global__ void k_tb1(const float* __restrict__ x,
                      const float* __restrict__ w1, const float* __restrict__ b1,
                      const float* __restrict__ w2, const float* __restrict__ b2,
                      const float* __restrict__ w3, const float* __restrict__ b3,
                      float* __restrict__ xs) {
    int idx = blockIdx.x * blockDim.x + threadIdx.x;
    if (idx >= T1 * BN * HID) return;
    int c    = idx & (HID - 1);
    int tmp  = idx >> 6;
    int node = tmp & (BN - 1);
    int t    = tmp >> 13;
    const float* xp = x + (node * HIS + t) * MSG;   // window x[t..t+2][0..1]
    float y1 = b1[c], y2 = b2[c], y3 = b3[c];
#pragma unroll
    for (int k = 0; k < 3; ++k) {
        float x0 = xp[k * MSG + 0];
        float x1 = xp[k * MSG + 1];
        y1 += w1[c * 6 + k] * x0 + w1[c * 6 + 3 + k] * x1;
        y2 += w2[c * 6 + k] * x0 + w2[c * 6 + 3 + k] * x1;
        y3 += w3[c * 6 + k] * x0 + w3[c * 6 + 3 + k] * x1;
    }
    float v = sigmoidf_(y3) * (y1 + y2);
    xs[idx] = v > 0.f ? v : 0.f;
}

// ---------------- K2: GAT projection via WMMA: h(GM,96) = xs(GM,64) @ W(64,96) ----------
// One wave per 16-row M strip; A fragments held in registers and reused across all
// six 16-col N tiles (12 x v_wmma_f32_16x16x32_f16 per wave, A traffic cut 6x).
__global__ void k_gemm_wmma(const float* __restrict__ xs,
                            const float* __restrict__ w,
                            float* __restrict__ h) {
    int wave  = blockIdx.x * (blockDim.x >> 5) + (threadIdx.x >> 5);  // M strip, grid exact
    int lane  = threadIdx.x & 31;
    const int khalf = lane >> 4;          // lane group selects K sub-range
    const int mlow  = lane & 15;
    const int mrow  = wave * 16 + mlow;
    const float* arow = xs + (size_t)mrow * HID;

    // A fragment layout (16-bit A 16x32): slots 0..7  -> K = khalf*8 + 0..7
    //                                     slots 8..15 -> K = 16 + khalf*8 + 0..7
    HFrag a0, a1;
    load_run8(arow + khalf * 8,          a0.h);
    load_run8(arow + 16 + khalf * 8,     a0.h + 8);
    load_run8(arow + 32 + khalf * 8,     a1.h);
    load_run8(arow + 48 + khalf * 8,     a1.h + 8);

    const float* brow0 = w + mlow;                      // N col = mlow (chunk 0)
    const float* brow1 = w + (size_t)32 * GN + mlow;    // N col = mlow, K += 32 (chunk 1)

#pragma unroll
    for (int nt = 0; nt < GN / 16; ++nt) {
        const int nb = nt * 16;
        // B fragment: lane = N col, slots 0..7 -> K = khalf*8 + s, 8..15 -> K = 16 + khalf*8 + s
        HFrag b0, b1;
#pragma unroll
        for (int s = 0; s < 8; ++s) {
            b0.h[s]     = (_Float16)brow0[(size_t)(khalf * 8 + s) * GN + nb];
            b0.h[8 + s] = (_Float16)brow0[(size_t)(16 + khalf * 8 + s) * GN + nb];
            b1.h[s]     = (_Float16)brow1[(size_t)(khalf * 8 + s) * GN + nb];
            b1.h[8 + s] = (_Float16)brow1[(size_t)(16 + khalf * 8 + s) * GN + nb];
        }

        CFrag c{};
        c.v = __builtin_amdgcn_wmma_f32_16x16x32_f16(false, a0.v, false, b0.v,
                                                     (short)0, c.v, false, false);
        c.v = __builtin_amdgcn_wmma_f32_16x16x32_f16(false, a1.v, false, b1.v,
                                                     (short)0, c.v, false, false);

#pragma unroll
        for (int v = 0; v < 8; ++v) {
            int m = wave * 16 + v + 8 * khalf;   // C/D layout: VGPR v -> M=v (+8 hi lanes)
            h[(size_t)m * GN + nb + mlow] = c.f[v];
        }
    }
}

// ---------------- K3: per-(row,head) attention coefficients; init max/sum ----------------
__global__ void k_att(const float* __restrict__ h,
                      const float* __restrict__ att_src, const float* __restrict__ att_dst,
                      float* __restrict__ a_src, float* __restrict__ a_dst,
                      float* __restrict__ mbuf, float* __restrict__ sbuf) {
    int i = blockIdx.x * blockDim.x + threadIdx.x;
    if (i >= GM * HEADS) return;
    int hd  = i % HEADS;
    int row = i / HEADS;
    const float* hp = h + (size_t)row * GN + hd * META;
    float s0 = 0.f, s1 = 0.f;
#pragma unroll
    for (int j = 0; j < META; ++j) {
        float v = hp[j];
        s0 += v * att_src[hd * META + j];
        s1 += v * att_dst[hd * META + j];
    }
    a_src[i] = s0;
    a_dst[i] = s1;
    mbuf[i]  = -3.0e38f;   // -inf for segment max
    sbuf[i]  = 0.f;
}

__global__ void k_zero(float* __restrict__ p, int n) {
    int i = blockIdx.x * blockDim.x + threadIdx.x;
    if (i < n) p[i] = 0.f;
}

// ---------------- edge helpers ----------------
__device__ __forceinline__ void edge_sd(const int* __restrict__ ei, int e, int& s, int& d) {
    if (e < E_) { s = ei[e]; d = ei[E_ + e]; }
    else        { s = d = e - E_; }           // self loop
}

// pass 1: segment max of leaky_relu(a_src[src]+a_dst[dst]) per (dst, head)
__global__ void k_edge_max(const int* __restrict__ ei,
                           const float* __restrict__ a_src, const float* __restrict__ a_dst,
                           float* __restrict__ mbuf) {
    int e = blockIdx.x * blockDim.x + threadIdx.x;
    if (e >= EP) return;
    int t = blockIdx.y;
    int s, d; edge_sd(ei, e, s, d);
    int sb = (t * BN + s) * HEADS, db = (t * BN + d) * HEADS;
#pragma unroll
    for (int hd = 0; hd < HEADS; ++hd) {
        float ev = a_src[sb + hd] + a_dst[db + hd];
        ev = ev > 0.f ? ev : NEG_SLOPE * ev;
        atomicMaxFloat(&mbuf[db + hd], ev);
    }
}

// pass 2: segment sum of exp(e - max)
__global__ void k_edge_sum(const int* __restrict__ ei,
                           const float* __restrict__ a_src, const float* __restrict__ a_dst,
                           const float* __restrict__ mbuf, float* __restrict__ sbuf) {
    int e = blockIdx.x * blockDim.x + threadIdx.x;
    if (e >= EP) return;
    int t = blockIdx.y;
    int s, d; edge_sd(ei, e, s, d);
    int sb = (t * BN + s) * HEADS, db = (t * BN + d) * HEADS;
#pragma unroll
    for (int hd = 0; hd < HEADS; ++hd) {
        float ev = a_src[sb + hd] + a_dst[db + hd];
        ev = ev > 0.f ? ev : NEG_SLOPE * ev;
        atomicAdd(&sbuf[db + hd], expf(ev - mbuf[db + hd]));
    }
}

// pass 3: out[dst] += sum_heads h[src,head]*alpha[head]/3  (head-mean folded in)
__global__ void k_edge_acc(const int* __restrict__ ei,
                           const float* __restrict__ a_src, const float* __restrict__ a_dst,
                           const float* __restrict__ mbuf, const float* __restrict__ sbuf,
                           const float* __restrict__ h, float* __restrict__ sp) {
    int e = blockIdx.x * blockDim.x + threadIdx.x;
    if (e >= EP) return;
    int t = blockIdx.y;
    int s, d; edge_sd(ei, e, s, d);
    int sb = (t * BN + s) * HEADS, db = (t * BN + d) * HEADS;
    float al[HEADS];
#pragma unroll
    for (int hd = 0; hd < HEADS; ++hd) {
        float ev = a_src[sb + hd] + a_dst[db + hd];
        ev = ev > 0.f ? ev : NEG_SLOPE * ev;
        al[hd] = expf(ev - mbuf[db + hd]) / (sbuf[db + hd] + 1e-16f) * (1.f / 3.f);
    }
    const float* hp = h + (size_t)(t * BN + s) * GN;  // 96 contiguous floats, 16B aligned
    float* op = sp + (size_t)(t * BN + d) * META;
#pragma unroll
    for (int c4 = 0; c4 < META / 4; ++c4) {
        float4 v0 = *(const float4*)(hp + c4 * 4);            // head 0
        float4 v1 = *(const float4*)(hp + META + c4 * 4);     // head 1
        float4 v2 = *(const float4*)(hp + 2 * META + c4 * 4); // head 2
        atomicAdd(&op[c4 * 4 + 0], v0.x * al[0] + v1.x * al[1] + v2.x * al[2]);
        atomicAdd(&op[c4 * 4 + 1], v0.y * al[0] + v1.y * al[1] + v2.y * al[2]);
        atomicAdd(&op[c4 * 4 + 2], v0.z * al[0] + v1.z * al[1] + v2.z * al[2]);
        atomicAdd(&op[c4 * 4 + 3], v0.w * al[0] + v1.w * al[1] + v2.w * al[2]);
    }
}

// bias + ReLU on GAT output
__global__ void k_sp_post(float* __restrict__ sp, const float* __restrict__ bias) {
    int i = blockIdx.x * blockDim.x + threadIdx.x;
    if (i >= GM * META) return;
    float v = sp[i] + bias[i & (META - 1)];
    sp[i] = v > 0.f ? v : 0.f;
}

// ---------------- K4: TimeBlock2 -> mk (node, t2), relu ----------------
__global__ void k_tb2(const float* __restrict__ sp,
                      const float* __restrict__ w1, const float* __restrict__ b1,
                      const float* __restrict__ w2, const float* __restrict__ b2,
                      const float* __restrict__ w3, const float* __restrict__ b3,
                      float* __restrict__ mk) {
    int i = blockIdx.x * blockDim.x + threadIdx.x;
    if (i >= BN * T2) return;
    int t2   = i % T2;
    int node = i / T2;
    float y1 = b1[0], y2 = b2[0], y3 = b3[0];
#pragma unroll
    for (int k = 0; k < 3; ++k) {
        const float* spp = sp + (size_t)((t2 + k) * BN + node) * META;
#pragma unroll
        for (int c = 0; c < META; ++c) {
            float v = spp[c];
            y1 += w1[c * 3 + k] * v;
            y2 += w2[c * 3 + k] * v;
            y3 += w3[c * 3 + k] * v;
        }
    }
    float v = sigmoidf_(y3) * (y1 + y2);
    mk[node * T2 + t2] = v > 0.f ? v : 0.f;
}

// ---------------- K5: final linear (8192,10) @ (10,32) + b ----------------
__global__ void k_mk(const float* __restrict__ mk,
                     const float* __restrict__ w, const float* __restrict__ b,
                     float* __restrict__ out) {
    int i = blockIdx.x * blockDim.x + threadIdx.x;
    if (i >= BN * META) return;
    int j    = i & (META - 1);
    int node = i >> 5;
    float acc = b[j];
#pragma unroll
    for (int t = 0; t < T2; ++t)
        acc += mk[node * T2 + t] * w[t * META + j];
    out[i] = acc;
}

// ---------------- launch ----------------
extern "C" void kernel_launch(void* const* d_in, const int* in_sizes, int n_in,
                              void* d_out, int out_size, void* d_ws, size_t ws_size,
                              hipStream_t stream) {
    const float* x       = (const float*)d_in[0];
    const int*   ei      = (const int*)  d_in[1];
    const float* tb1_w1  = (const float*)d_in[2];
    const float* tb1_w2  = (const float*)d_in[3];
    const float* tb1_w3  = (const float*)d_in[4];
    const float* tb1_b1  = (const float*)d_in[5];
    const float* tb1_b2  = (const float*)d_in[6];
    const float* tb1_b3  = (const float*)d_in[7];
    const float* lin_w   = (const float*)d_in[8];
    const float* att_src = (const float*)d_in[9];
    const float* att_dst = (const float*)d_in[10];
    const float* g_bias  = (const float*)d_in[11];
    const float* tb2_w1  = (const float*)d_in[12];
    const float* tb2_w2  = (const float*)d_in[13];
    const float* tb2_w3  = (const float*)d_in[14];
    const float* tb2_b1  = (const float*)d_in[15];
    const float* tb2_b2  = (const float*)d_in[16];
    const float* tb2_b3  = (const float*)d_in[17];
    const float* mk_w    = (const float*)d_in[18];
    const float* mk_b    = (const float*)d_in[19];

    // workspace layout (floats)
    float* ws    = (float*)d_ws;
    float* xs    = ws;                               // GM*HID   = 6,291,456
    float* h     = xs    + (size_t)GM * HID;         // GM*GN    = 9,437,184
    float* a_src = h     + (size_t)GM * GN;          // GM*HEADS
    float* a_dst = a_src + (size_t)GM * HEADS;
    float* mbuf  = a_dst + (size_t)GM * HEADS;
    float* sbuf  = mbuf  + (size_t)GM * HEADS;
    float* sp    = sbuf  + (size_t)GM * HEADS;       // GM*META  = 3,145,728
    float* mk    = sp    + (size_t)GM * META;        // BN*T2

    const int TB = 256;

    // 1) TimeBlock1 -> xs (t, node, c)
    k_tb1<<<(T1 * BN * HID + TB - 1) / TB, TB, 0, stream>>>(
        x, tb1_w1, tb1_b1, tb1_w2, tb1_b2, tb1_w3, tb1_b3, xs);

    // 2) WMMA GEMM: h = xs @ lin_w   (one wave / 16-row strip, 8 waves / block)
    {
        int strips = GM / 16;                        // 6144 waves
        k_gemm_wmma<<<strips / 8, TB, 0, stream>>>(xs, lin_w, h);
    }

    // 3) attention coefficients + init max/sum
    k_att<<<(GM * HEADS + TB - 1) / TB, TB, 0, stream>>>(
        h, att_src, att_dst, a_src, a_dst, mbuf, sbuf);

    // zero GAT accumulators
    k_zero<<<(GM * META + TB - 1) / TB, TB, 0, stream>>>(sp, GM * META);

    // 4) edge softmax + scatter (3 passes), all 12 timesteps via grid.y
    dim3 egrid((EP + TB - 1) / TB, T1);
    k_edge_max<<<egrid, TB, 0, stream>>>(ei, a_src, a_dst, mbuf);
    k_edge_sum<<<egrid, TB, 0, stream>>>(ei, a_src, a_dst, mbuf, sbuf);
    k_edge_acc<<<egrid, TB, 0, stream>>>(ei, a_src, a_dst, mbuf, sbuf, h, sp);

    // 5) bias + relu
    k_sp_post<<<(GM * META + TB - 1) / TB, TB, 0, stream>>>(sp, g_bias);

    // 6) TimeBlock2 -> mk
    k_tb2<<<(BN * T2 + TB - 1) / TB, TB, 0, stream>>>(
        sp, tb2_w1, tb2_b1, tb2_w2, tb2_b2, tb2_w3, tb2_b3, mk);

    // 7) final linear -> d_out
    k_mk<<<(BN * META + TB - 1) / TB, TB, 0, stream>>>(
        mk, mk_w, mk_b, (float*)d_out);
}
